// RecurrentCNN_43473658970125
// MI455X (gfx1250) — matrix operand order repaired and checked by reading, not compiled, not run on hardware
//
#include <hip/hip_runtime.h>
#include <hip/hip_bf16.h>

typedef unsigned short ushort_t;
typedef __attribute__((ext_vector_type(8)))  float  v8f;
typedef __attribute__((ext_vector_type(8)))  __bf16 bf16x8;
typedef __attribute__((ext_vector_type(16))) __bf16 bf16x16;
typedef __attribute__((ext_vector_type(4)))  int    v4i;

#define BN_S_D 0.9999950000374997f   // 1/sqrt(1+1e-5)

#if __has_builtin(__builtin_amdgcn_global_load_async_to_lds_b128) && \
    __has_builtin(__builtin_amdgcn_s_wait_asynccnt)
#define USE_ASYNC_LDS 1
typedef __attribute__((address_space(1))) v4i* g4_ptr_t;   // global v4i*
typedef __attribute__((address_space(3))) v4i* l4_ptr_t;   // LDS v4i*
#else
#define USE_ASYNC_LDS 0
#endif

__device__ __forceinline__ float bf2f(ushort_t u) {
    unsigned x = ((unsigned)u) << 16;
    float f; __builtin_memcpy(&f, &x, 4); return f;
}
__device__ __forceinline__ ushort_t f2bf(float f) {
    unsigned x; __builtin_memcpy(&x, &f, 4);
    unsigned r = x + 0x7FFFu + ((x >> 16) & 1u);   // RNE
    return (ushort_t)(r >> 16);
}
__device__ __forceinline__ float sigm(float x) { return 1.f / (1.f + __expf(-x)); }

// ---------------------------------------------------------------------------
// f32 -> bf16 weight conversion
// ---------------------------------------------------------------------------
__global__ void cvt_bf16_kernel(const float* __restrict__ src, ushort_t* __restrict__ dst, int n) {
    int i = blockIdx.x * blockDim.x + threadIdx.x;
    if (i < n) dst[i] = f2bf(src[i]);
}

// ---------------------------------------------------------------------------
// Preprocess: channel0 = x, channel1 = masked-log10 normalized (per sample)
// ---------------------------------------------------------------------------
__global__ void preprocess_kernel(const float* __restrict__ x, float* __restrict__ x2) {
    const int T = 1024;
    int b = blockIdx.x, tid = threadIdx.x;
    __shared__ float red[256];
    float lmin = 3.0e38f;
    for (int t = tid; t < T; t += 256) {
        float v = x[b * T + t];
        if (v >= 1e-4f) lmin = fminf(lmin, log10f(v));
    }
    red[tid] = lmin; __syncthreads();
    for (int s = 128; s > 0; s >>= 1) { if (tid < s) red[tid] = fminf(red[tid], red[tid + s]); __syncthreads(); }
    float mn = red[0]; __syncthreads();
    float lmax = -3.0e38f;
    for (int t = tid; t < T; t += 256) {
        float v = x[b * T + t];
        if (v >= 1e-4f) lmax = fmaxf(lmax, log10f(v) - mn);
    }
    red[tid] = lmax; __syncthreads();
    for (int s = 128; s > 0; s >>= 1) { if (tid < s) red[tid] = fmaxf(red[tid], red[tid + s]); __syncthreads(); }
    float mx = fmaxf(red[0], 1e-12f);
    for (int t = tid; t < T; t += 256) {
        float v = x[b * T + t];
        float o = (v >= 1e-4f) ? (log10f(v) - mn) / mx : 0.f;
        x2[(b * 2 + 0) * T + t] = v;
        x2[(b * 2 + 1) * T + t] = o;
    }
}

// ---------------------------------------------------------------------------
// Direct conv1d (K=5, pad=2) + eval-BN scale + SiLU. Layout [B,C,T].
// ---------------------------------------------------------------------------
__global__ void conv_silu_kernel(const float* __restrict__ in, const float* __restrict__ w,
                                 const float* __restrict__ bias, float* __restrict__ out,
                                 int Cin, int Cout) {
    const int T = 1024;
    int idx = blockIdx.x * blockDim.x + threadIdx.x;
    int total = 16 * Cout * T;
    if (idx >= total) return;
    int t = idx % T, co = (idx / T) % Cout, b = idx / (T * Cout);
    float acc = bias[co];
    for (int ci = 0; ci < Cin; ci++) {
        const float* ip = in + ((size_t)b * Cin + ci) * T;
        const float* wp = w + ((size_t)co * Cin + ci) * 5;
#pragma unroll
        for (int k = 0; k < 5; k++) {
            int tt = t + k - 2;
            if (tt >= 0 && tt < T) acc += ip[tt] * wp[k];
        }
    }
    float z = acc * BN_S_D;
    out[idx] = z * sigm(z);
}

// [B,128,T] f32 -> [B*T,128] bf16
__global__ void to_seq_kernel(const float* __restrict__ in, ushort_t* __restrict__ seq) {
    int idx = blockIdx.x * blockDim.x + threadIdx.x;
    if (idx >= 16 * 128 * 1024) return;
    int c = idx & 127, t = (idx >> 7) & 1023, b = idx >> 17;
    seq[idx] = f2bf(in[((size_t)b * 128 + c) * 1024 + t]);
}

// ---------------------------------------------------------------------------
// Generic WMMA GEMM: Out[M,N](bf16) = act( A[M,K](bf16) @ W[N,K]^T + b1 + b2 )
// 4 waves/block, wave owns a 16 x (16*JTILES) tile.  Per k-chunk: all B
// fragments are loaded into distinct registers first, then all WMMAs issue —
// lets the scheduler stagger s_wait_loadcnt and overlap loads with the
// matrix pipe.  tmajor: A row r maps to seq row (r&15)*Tseq + (r>>4).
// ---------------------------------------------------------------------------
template <int JTILES>
__global__ __launch_bounds__(128)
void gemm_bf16_kernel(const ushort_t* __restrict__ A, const ushort_t* __restrict__ W,
                      const float* __restrict__ b1, const float* __restrict__ b2,
                      ushort_t* __restrict__ Out,
                      int M, int N, int K, int act, int tmajor, int Tseq) {
    int wave = threadIdx.x >> 5, lane = threadIdx.x & 31;
    int nl = lane & 15, g = lane >> 4;
    int mBase = (blockIdx.y * 4 + wave) * 16;
    int nBase = blockIdx.x * (16 * JTILES);
    if (mBase >= M) return;
    v8f acc[JTILES] = {};
    int row = mBase + nl;
    long arow = tmajor ? (long)(row & 15) * Tseq + (row >> 4) : (long)row;
    const ushort_t* Arow = A + arow * (long)K;
    int KC = K >> 5;
    for (int kc = 0; kc < KC; ++kc) {
        int k0 = kc << 5;
        __builtin_prefetch(Arow + k0 + 64, 0, 3);   // global_prefetch_b8, near-cache
        bf16x8 lo = *(const bf16x8*)(Arow + k0 + g * 16);
        bf16x8 hi = *(const bf16x8*)(Arow + k0 + g * 16 + 8);
        bf16x16 af;
#pragma unroll
        for (int e = 0; e < 8; e++) { af[e] = lo[e]; af[e + 8] = hi[e]; }
        bf16x16 bfr[JTILES];
#pragma unroll
        for (int j = 0; j < JTILES; j++)
            bfr[j] = *(const bf16x16*)(W + (long)(nBase + j * 16 + nl) * K + k0 + g * 16);
#pragma unroll
        for (int j = 0; j < JTILES; j++)
            acc[j] = __builtin_amdgcn_wmma_f32_16x16x32_bf16(false, af, false, bfr[j],
                                                             (short)0, acc[j], false, false);
    }
#pragma unroll
    for (int j = 0; j < JTILES; j++) {
        int col = nBase + j * 16 + nl;
        float bias = (b1 ? b1[col] : 0.f) + (b2 ? b2[col] : 0.f);
#pragma unroll
        for (int r = 0; r < 8; r++) {
            int orow = mBase + 8 * g + r;
            float v = acc[j][r] + bias;
            if (act) { float z = v * BN_S_D; v = z * sigm(z); }
            Out[(long)orow * N + col] = f2bf(v);
        }
    }
}

// ---------------------------------------------------------------------------
// LSTM recurrence scan. One block per direction (blockIdx.x), 8 waves.
// Wave w owns hidden-unit slice [w*H/8, (w+1)*H/8) in ALL four gate blocks,
// so i/f/g/o land in its own WMMA accumulators and c stays in VGPRs.
// h lives in LDS (padded rows -> conflict-free ds_load_b128 A-fragments).
// For H=128, Whh (128 KB) is staged once into LDS via the gfx1250 async
// global->LDS path, so 1024 steps of B-fragments come from LDS.
// Per k-chunk all 4*JT B fragments load first, then all WMMAs (overlap).
// xg is t-major: [dir][t][16 batch][4H] bf16 (biases folded in).
// ---------------------------------------------------------------------------
template <int H>
__global__ __launch_bounds__(256)
void lstm_scan_kernel(const ushort_t* __restrict__ xg,
                      const ushort_t* __restrict__ whh_f, const ushort_t* __restrict__ whh_b,
                      ushort_t* __restrict__ seqOut, float* __restrict__ h0_out, int T) {
    constexpr int JT = H / 128;      // j-tiles per wave (1 for H=128, 2 for H=256)
    constexpr int KC = H / 32;
    constexpr int HS = H + 8;        // padded row stride (bank spread)
    constexpr bool STAGE = (H == 128);
    const int N4 = 4 * H;
    const int dir = blockIdx.x;
    const ushort_t* whh = dir ? whh_b : whh_f;
    const ushort_t* xgd = xg + (size_t)dir * T * 16 * N4;

    extern __shared__ char smem[];
    ushort_t* hb = (ushort_t*)smem;                                  // [16][HS]
    ushort_t* wl = (ushort_t*)(smem + ((16 * HS * 2 + 63) & ~63));   // [4H][HS] (STAGE only)

    int wave = threadIdx.x >> 5, lane = threadIdx.x & 31;
    int nl = lane & 15, g = lane >> 4;
    int j0w = wave * (H / 8);

    if (STAGE) {
        const int CPR = H / 8;                       // 16B chunks per row
        for (int i = threadIdx.x; i < 4 * H * CPR; i += 256) {
            int rr = i / CPR, cc = (i % CPR) * 8;
#if USE_ASYNC_LDS
            __builtin_amdgcn_global_load_async_to_lds_b128(
                (g4_ptr_t)(ushort_t*)(whh + (size_t)rr * H + cc),
                (l4_ptr_t)(wl + (size_t)rr * HS + cc), 0, 0);
#else
            *(bf16x8*)(wl + (size_t)rr * HS + cc) =
                *(const bf16x8*)(whh + (size_t)rr * H + cc);
#endif
        }
#if USE_ASYNC_LDS
        __builtin_amdgcn_s_wait_asynccnt(0);
#endif
    }
    for (int i = threadIdx.x; i < 16 * HS; i += 256) hb[i] = 0;
    float cst[JT][8];
#pragma unroll
    for (int jt = 0; jt < JT; jt++)
#pragma unroll
        for (int r = 0; r < 8; r++) cst[jt][r] = 0.f;
    __syncthreads();

    for (int step = 0; step < T; ++step) {
        int t = dir ? (T - 1 - step) : step;
        v8f acc[JT][4];
        const ushort_t* xrow0 = xgd + (size_t)(t * 16) * N4;
#pragma unroll
        for (int jt = 0; jt < JT; jt++)
#pragma unroll
            for (int q = 0; q < 4; q++) {
                int col = q * H + j0w + jt * 16 + nl;
#pragma unroll
                for (int r = 0; r < 8; r++)
                    acc[jt][q][r] = bf2f(xrow0[(size_t)(8 * g + r) * N4 + col]);
            }
#pragma unroll
        for (int kc = 0; kc < KC; kc++) {
            int k0 = kc * 32;
            bf16x8 lo = *(const bf16x8*)(hb + nl * HS + k0 + g * 16);
            bf16x8 hi = *(const bf16x8*)(hb + nl * HS + k0 + g * 16 + 8);
            bf16x16 af;
#pragma unroll
            for (int e = 0; e < 8; e++) { af[e] = lo[e]; af[e + 8] = hi[e]; }
            bf16x16 barr[JT][4];
            if (STAGE) {
#pragma unroll
                for (int jt = 0; jt < JT; jt++)
#pragma unroll
                    for (int q = 0; q < 4; q++) {
                        int col0 = q * H + j0w + jt * 16;
                        bf16x8 bl = *(const bf16x8*)(wl + (size_t)(col0 + nl) * HS + k0 + g * 16);
                        bf16x8 bh = *(const bf16x8*)(wl + (size_t)(col0 + nl) * HS + k0 + g * 16 + 8);
#pragma unroll
                        for (int e = 0; e < 8; e++) { barr[jt][q][e] = bl[e]; barr[jt][q][e + 8] = bh[e]; }
                    }
            } else {
#pragma unroll
                for (int jt = 0; jt < JT; jt++)
#pragma unroll
                    for (int q = 0; q < 4; q++) {
                        int col0 = q * H + j0w + jt * 16;
                        barr[jt][q] = *(const bf16x16*)(whh + (size_t)(col0 + nl) * H + k0 + g * 16);
                    }
            }
#pragma unroll
            for (int jt = 0; jt < JT; jt++)
#pragma unroll
                for (int q = 0; q < 4; q++)
                    acc[jt][q] = __builtin_amdgcn_wmma_f32_16x16x32_bf16(false, af, false, barr[jt][q],
                                                                         (short)0, acc[jt][q], false, false);
        }
        __syncthreads();   // all reads of hb for this step done
#pragma unroll
        for (int jt = 0; jt < JT; jt++) {
            int j = j0w + jt * 16 + nl;
#pragma unroll
            for (int r = 0; r < 8; r++) {
                int bb = 8 * g + r;
                float iv = sigm(acc[jt][0][r]);
                float fv = sigm(acc[jt][1][r]);
                float gv = tanhf(acc[jt][2][r]);
                float ov = sigm(acc[jt][3][r]);
                float c = fv * cst[jt][r] + iv * gv;
                cst[jt][r] = c;
                float h = ov * tanhf(c);
                ushort_t hbv = f2bf(h);
                hb[bb * HS + j] = hbv;
                seqOut[((size_t)bb * T + t) * (2 * H) + (size_t)dir * H + j] = hbv;
                if (h0_out && dir == 0 && step == T - 1) h0_out[bb * H + j] = h;
            }
        }
        __syncthreads();   // new h visible before next step
    }
}

// ---------------------------------------------------------------------------
// cls MLP head: [16,256] -> 2, all in one block (tiny).
// ---------------------------------------------------------------------------
__global__ void cls_mlp_kernel(const float* __restrict__ h0,
                               const float* w0, const float* b0, const float* w1, const float* b1,
                               const float* w2, const float* b2, const float* w3, const float* b3,
                               const float* w4, const float* b4, const float* w5, const float* b5,
                               const float* w6, const float* b6, float* __restrict__ out) {
    __shared__ float bufA[16 * 256];
    __shared__ float bufB[16 * 128];
    const float* Ws[7] = {w0, w1, w2, w3, w4, w5, w6};
    const float* Bs[7] = {b0, b1, b2, b3, b4, b5, b6};
    const int dims[8] = {256, 128, 64, 32, 16, 8, 4, 2};
    for (int i = threadIdx.x; i < 16 * 256; i += blockDim.x) bufA[i] = h0[i];
    __syncthreads();
    float* cur = bufA; float* nxt = bufB;
    for (int L = 0; L < 7; L++) {
        int din = dims[L], dout = dims[L + 1];
        const float* W = Ws[L]; const float* Bb = Bs[L];
        for (int idx = threadIdx.x; idx < 16 * dout; idx += blockDim.x) {
            int bb = idx / dout, n = idx % dout;
            float s = Bb[n];
            for (int k = 0; k < din; k++) s += cur[bb * din + k] * W[n * din + k];
            if (L < 6) { float z = s * BN_S_D; s = z * sigm(z); }
            if (L == 6) out[bb * 2 + n] = s; else nxt[bb * dout + n] = s;
        }
        __syncthreads();
        float* tmp = cur; cur = nxt; nxt = tmp;
    }
}

// ---------------------------------------------------------------------------
// int MLP tail: 16 -> 8 -> 4 -> 2 per row, writes transposed [B,2,T].
// ---------------------------------------------------------------------------
__global__ void int_tail_kernel(const ushort_t* __restrict__ in,
                                const float* __restrict__ w5, const float* __restrict__ b5,
                                const float* __restrict__ w6, const float* __restrict__ b6,
                                const float* __restrict__ w7, const float* __restrict__ b7,
                                float* __restrict__ out) {
    int r = blockIdx.x * blockDim.x + threadIdx.x;
    if (r >= 16384) return;
    float a[16];
#pragma unroll
    for (int k = 0; k < 16; k++) a[k] = bf2f(in[(size_t)r * 16 + k]);
    float h8[8];
#pragma unroll
    for (int n = 0; n < 8; n++) {
        float s = b5[n];
#pragma unroll
        for (int k = 0; k < 16; k++) s += a[k] * w5[n * 16 + k];
        float z = s * BN_S_D; h8[n] = z * sigm(z);
    }
    float h4[4];
#pragma unroll
    for (int n = 0; n < 4; n++) {
        float s = b6[n];
#pragma unroll
        for (int k = 0; k < 8; k++) s += h8[k] * w6[n * 8 + k];
        float z = s * BN_S_D; h4[n] = z * sigm(z);
    }
    int b = r >> 10, t = r & 1023;
#pragma unroll
    for (int n = 0; n < 2; n++) {
        float s = b7[n];
#pragma unroll
        for (int k = 0; k < 4; k++) s += h4[k] * w7[n * 4 + k];
        out[((size_t)b * 2 + n) * 1024 + t] = s;   // transposed [B,2,T]
    }
}

// ---------------------------------------------------------------------------
// Host orchestration
// ---------------------------------------------------------------------------
extern "C" void kernel_launch(void* const* d_in, const int* in_sizes, int n_in,
                              void* d_out, int out_size, void* d_ws, size_t ws_size,
                              hipStream_t stream) {
    (void)in_sizes; (void)n_in; (void)out_size; (void)ws_size;
    const int B = 16, T = 1024, M = B * T;

    char* base = (char*)d_ws; size_t off = 0;
    auto alloc = [&](size_t bytes) -> char* {
        char* p = base + off; off = (off + bytes + 255) & ~(size_t)255; return p;
    };
    float*    x2   = (float*)alloc((size_t)B * 2 * T * 4);
    float*    cA   = (float*)alloc((size_t)B * 128 * T * 4);
    float*    cB   = (float*)alloc((size_t)B * 128 * T * 4);
    ushort_t* seqP = (ushort_t*)alloc((size_t)M * 512 * 2);
    ushort_t* seqQ = (ushort_t*)alloc((size_t)M * 512 * 2);
    ushort_t* xg   = (ushort_t*)alloc((size_t)2 * M * 1024 * 2);
    float*    h0   = (float*)alloc((size_t)16 * 256 * 4);

    // bf16 weight conversion (LSTM wih/whh, int-MLP first 5 layers)
    ushort_t* wihC[2][5][2]; ushort_t* whhC[2][5][2];
    for (int s = 0; s < 2; s++) {
        int H = s ? 256 : 128, baseIdx = s ? 53 : 13;
        for (int l = 0; l < 5; l++)
            for (int d = 0; d < 2; d++) {
                int K = l ? 2 * H : (s ? 256 : 128);
                int idx = baseIdx + l * 8 + d * 4;
                int nwih = 4 * H * K, nwhh = 4 * H * H;
                wihC[s][l][d] = (ushort_t*)alloc((size_t)nwih * 2);
                whhC[s][l][d] = (ushort_t*)alloc((size_t)nwhh * 2);
                cvt_bf16_kernel<<<(nwih + 255) / 256, 256, 0, stream>>>((const float*)d_in[idx], wihC[s][l][d], nwih);
                cvt_bf16_kernel<<<(nwhh + 255) / 256, 256, 0, stream>>>((const float*)d_in[idx + 1], whhC[s][l][d], nwhh);
            }
    }
    const int idims[6] = {512, 256, 128, 64, 32, 16};
    ushort_t* intW[5];
    for (int i = 0; i < 5; i++) {
        int n = idims[i + 1] * idims[i];
        intW[i] = (ushort_t*)alloc((size_t)n * 2);
        cvt_bf16_kernel<<<(n + 255) / 256, 256, 0, stream>>>((const float*)d_in[107 + 2 * i], intW[i], n);
    }

    // branch-free GEMM dispatch by N-tile multiple
    auto gemm = [&](const ushort_t* Ap, const ushort_t* Wp, const float* bb1, const float* bb2,
                    ushort_t* Op, int N, int K, int act, int tmajor) {
        dim3 gy(1, (M + 63) / 64);
        if (N % 64 == 0) { gy.x = N / 64; gemm_bf16_kernel<4><<<gy, 128, 0, stream>>>(Ap, Wp, bb1, bb2, Op, M, N, K, act, tmajor, T); }
        else if (N % 32 == 0) { gy.x = N / 32; gemm_bf16_kernel<2><<<gy, 128, 0, stream>>>(Ap, Wp, bb1, bb2, Op, M, N, K, act, tmajor, T); }
        else { gy.x = N / 16; gemm_bf16_kernel<1><<<gy, 128, 0, stream>>>(Ap, Wp, bb1, bb2, Op, M, N, K, act, tmajor, T); }
    };

    // preprocess + conv chain
    preprocess_kernel<<<B, 256, 0, stream>>>((const float*)d_in[0], x2);
    const int chans[7] = {2, 4, 8, 16, 32, 64, 128};
    const float* cin = x2; float* couts[2] = {cA, cB};
    for (int i = 0; i < 6; i++) {
        int total = B * chans[i + 1] * T;
        conv_silu_kernel<<<(total + 255) / 256, 256, 0, stream>>>(
            cin, (const float*)d_in[1 + 2 * i], (const float*)d_in[2 + 2 * i],
            couts[i & 1], chans[i], chans[i + 1]);
        cin = couts[i & 1];
    }
    { int total = B * 128 * T; to_seq_kernel<<<(total + 255) / 256, 256, 0, stream>>>(cin, seqP); }

    // dynamic LDS sizes for the scan kernels
    const size_t shmem128 = ((16 * (128 + 8) * 2 + 63) & ~(size_t)63) + (size_t)512 * (128 + 8) * 2;
    const size_t shmem256 = (size_t)16 * (256 + 8) * 2;

    // bidirectional LSTM stacks
    ushort_t* sIn = seqP; ushort_t* sOut = seqQ;
    for (int s = 0; s < 2; s++) {
        int H = s ? 256 : 128, baseIdx = s ? 53 : 13, N4 = 4 * H;
        for (int l = 0; l < 5; l++) {
            int K = l ? 2 * H : (s ? 256 : 128);
            int iF = baseIdx + l * 8, iB = iF + 4;
            gemm(sIn, wihC[s][l][0], (const float*)d_in[iF + 2], (const float*)d_in[iF + 3],
                 xg, N4, K, 0, 1);
            gemm(sIn, wihC[s][l][1], (const float*)d_in[iB + 2], (const float*)d_in[iB + 3],
                 xg + (size_t)M * N4, N4, K, 0, 1);
            float* h0p = (s == 1 && l == 0) ? h0 : nullptr;
            if (H == 128)
                lstm_scan_kernel<128><<<2, 256, shmem128, stream>>>(xg, whhC[s][l][0], whhC[s][l][1], sOut, h0p, T);
            else
                lstm_scan_kernel<256><<<2, 256, shmem256, stream>>>(xg, whhC[s][l][0], whhC[s][l][1], sOut, h0p, T);
            ushort_t* tmp = sIn; sIn = sOut; sOut = tmp;
        }
    }

    // int MLP: WMMA layers (K >= 32) with fused BN*SiLU
    for (int i = 0; i < 5; i++) {
        gemm(sIn, intW[i], (const float*)d_in[108 + 2 * i], nullptr, sOut, idims[i + 1], idims[i], 1, 0);
        ushort_t* tmp = sIn; sIn = sOut; sOut = tmp;
    }
    int_tail_kernel<<<(M + 255) / 256, 256, 0, stream>>>(sIn,
        (const float*)d_in[117], (const float*)d_in[118],
        (const float*)d_in[119], (const float*)d_in[120],
        (const float*)d_in[121], (const float*)d_in[122],
        (float*)d_out + 32);

    cls_mlp_kernel<<<1, 256, 0, stream>>>(h0,
        (const float*)d_in[93], (const float*)d_in[94], (const float*)d_in[95], (const float*)d_in[96],
        (const float*)d_in[97], (const float*)d_in[98], (const float*)d_in[99], (const float*)d_in[100],
        (const float*)d_in[101], (const float*)d_in[102], (const float*)d_in[103], (const float*)d_in[104],
        (const float*)d_in[105], (const float*)d_in[106], (float*)d_out);
}
